// MultiScaleAttention_77043123356274
// MI455X (gfx1250) — matrix operand order repaired and checked by reading, not compiled, not run on hardware
//
#include <hip/hip_runtime.h>
#include <stdint.h>

// ---------------------------------------------------------------------------
// Types
// ---------------------------------------------------------------------------
typedef unsigned short u16;
typedef __attribute__((ext_vector_type(16))) __bf16 v16bf;
typedef __attribute__((ext_vector_type(8)))  float  v8f;
typedef __attribute__((ext_vector_type(4)))  unsigned u32x4;

union FragU { u32x4 u[2]; v16bf v; };

__device__ __forceinline__ u16 f2bf(float f) {
  unsigned u = __float_as_uint(f);
  unsigned r = u + 0x7FFFu + ((u >> 16) & 1u);   // RNE
  return (u16)(r >> 16);
}
__device__ __forceinline__ float bf2f(u16 h) {
  return __uint_as_float(((unsigned)h) << 16);
}

// A-fragment (16x32 bf16): lanes 0-15 hold K 0..7 & 16..23; lanes 16-31 hold
// K 8..15 & 24..31 (ISA 7.12.2). Row-major source => two b128 loads.
__device__ __forceinline__ v16bf load_fragA(const u16* __restrict__ row, int kbase, int lane) {
  int of = (lane < 16) ? 0 : 8;
  FragU f;
  f.u[0] = *reinterpret_cast<const u32x4*>(row + kbase + of);
  f.u[1] = *reinterpret_cast<const u32x4*>(row + kbase + of + 16);
  return f.v;
}
// B-fragment (32x16 bf16): lane holds column N=lane&15; lanes 0-15 K 0..7 & 16..23,
// lanes 16-31 K 8..15 & 24..31. Source is Bt[n][k] (column-major B) row-major.
__device__ __forceinline__ v16bf load_fragB(const u16* __restrict__ rowN, int kbase, int lane) {
  int of = (lane < 16) ? 0 : 8;
  FragU f;
  f.u[0] = *reinterpret_cast<const u32x4*>(rowN + kbase + of);
  f.u[1] = *reinterpret_cast<const u32x4*>(rowN + kbase + of + 16);
  return f.v;
}

// ---------------------------------------------------------------------------
// Problem constants
// ---------------------------------------------------------------------------
#define BATCH 4
#define SPH 64
#define SPW 64
#define DIM 768
#define HEADS 12
#define HD 64
#define NTOK 4097          // 1 + 64*64
#define MQKV 16388         // BATCH * NTOK
#define MQKV_PAD 16400
#define LPOOL 1025         // 1 + 32*32
#define QROWS 1040         // 16 (cls pad) + 1024
#define KROWS 1056         // 32 (cls pad) + 1024
#define MPROJ 4100         // BATCH * LPOOL
#define MPROJ_PAD 4112

// ---------------------------------------------------------------------------
// Prep kernels
// ---------------------------------------------------------------------------
__global__ void f32_to_bf16_kernel(const float* __restrict__ src, u16* __restrict__ dst, long long n) {
  long long i = (long long)blockIdx.x * blockDim.x + threadIdx.x;
  if (i < n) dst[i] = f2bf(src[i]);
}

// dst[n*K + k] = src[k*N + n]   (bf16 transpose of a KxN fp32 matrix)
__global__ void transpose_bf16_kernel(const float* __restrict__ src, u16* __restrict__ dst, int K, int N) {
  long long i = (long long)blockIdx.x * blockDim.x + threadIdx.x;
  long long total = (long long)K * N;
  if (i >= total) return;
  int n = (int)(i / K);
  int k = (int)(i % K);
  dst[i] = f2bf(src[(long long)k * N + n]);
}

// ---------------------------------------------------------------------------
// qkv GEMM: [MQKV_PAD x 768] (bf16) @ Bt [2304 x 768] (bf16) -> bf16 scattered
// into layout [which][b][h][tok][c]. One wave computes a 16x64 tile.
// ---------------------------------------------------------------------------
__global__ __launch_bounds__(32)
void gemm_qkv_kernel(const u16* __restrict__ A, const u16* __restrict__ Bt,
                     u16* __restrict__ qkv_out, int Mvalid) {
  int lane = threadIdx.x;
  int m0 = blockIdx.x * 16;
  int n0 = blockIdx.y * 64;
  v8f acc[4] = {v8f{}, v8f{}, v8f{}, v8f{}};
  const u16* arow = A + (size_t)(m0 + (lane & 15)) * 768;
  for (int k = 0; k < 768; k += 32) {
    __builtin_prefetch((const void*)(arow + k + 128), 0, 1);
    v16bf a = load_fragA(arow, k, lane);
#pragma unroll
    for (int j = 0; j < 4; ++j) {
      const u16* brow = Bt + (size_t)(n0 + j * 16 + (lane & 15)) * 768;
      v16bf b = load_fragB(brow, k, lane);
      acc[j] = __builtin_amdgcn_wmma_f32_16x16x32_bf16(false, a, false, b, (short)0, acc[j], false, false);
    }
  }
#pragma unroll
  for (int j = 0; j < 4; ++j) {
#pragma unroll
    for (int r = 0; r < 8; ++r) {
      int row = m0 + r + ((lane < 16) ? 0 : 8);
      if (row >= Mvalid) continue;
      int col = n0 + j * 16 + (lane & 15);
      int b = row / NTOK, tok = row % NTOK;
      int which = col / 768, rem = col % 768;
      int h = rem >> 6, c = rem & 63;
      size_t idx = ((((size_t)which * BATCH + b) * HEADS + h) * NTOK + tok) * HD + c;
      qkv_out[idx] = f2bf(acc[j][r]);
    }
  }
}

// ---------------------------------------------------------------------------
// Pool (3x3 dwconv stride 2, SAME pad (0,1)) + LayerNorm over HD.
// One block = one (which, b, h, pooled-token), 64 threads = channels.
// Writes padded bf16 Q/K/Vt tiles + fp32 q for residual / rel-pos.
// ---------------------------------------------------------------------------
__global__ __launch_bounds__(64)
void pool_ln_kernel(const u16* __restrict__ qkv,
                    const float* __restrict__ dw_q, const float* __restrict__ dw_k,
                    const float* __restrict__ dw_v,
                    const float* __restrict__ gq, const float* __restrict__ bq,
                    const float* __restrict__ gk, const float* __restrict__ bk,
                    const float* __restrict__ gv, const float* __restrict__ bv,
                    u16* __restrict__ Qpad, float* __restrict__ qres,
                    u16* __restrict__ Kpad, u16* __restrict__ Vt) {
  int c = threadIdx.x;
  int blk = blockIdx.x;
  int t = blk % LPOOL; int rest = blk / LPOOL;
  int h = rest % HEADS; rest /= HEADS;
  int b = rest % BATCH; int which = rest / BATCH;

  const u16* base = qkv + (((size_t)which * BATCH + b) * HEADS + h) * (size_t)NTOK * HD;
  float val;
  if (t == 0) {
    val = bf2f(base[c]);
  } else {
    int p = t - 1, py = p >> 5, px = p & 31;
    const float* dw = (which == 0) ? dw_q : ((which == 1) ? dw_k : dw_v);
    float s = 0.f;
#pragma unroll
    for (int dy = 0; dy < 3; ++dy) {
      int iy = 2 * py + dy;
      if (iy >= SPH) continue;
#pragma unroll
      for (int dx = 0; dx < 3; ++dx) {
        int ix = 2 * px + dx;
        if (ix >= SPW) continue;
        float wv = dw[(dy * 3 + dx) * DIM + h * HD + c];
        float xv = bf2f(base[(size_t)(1 + iy * SPW + ix) * HD + c]);
        s += wv * xv;
      }
    }
    val = s;
  }

  __shared__ float rs[64], rq[64];
  rs[c] = val; rq[c] = val * val;
  __syncthreads();
  for (int sdiv = 32; sdiv > 0; sdiv >>= 1) {
    if (c < sdiv) { rs[c] += rs[c + sdiv]; rq[c] += rq[c + sdiv]; }
    __syncthreads();
  }
  float mean = rs[0] * (1.f / 64.f);
  float var  = rq[0] * (1.f / 64.f) - mean * mean;
  const float* g; const float* be;
  if (which == 0)      { g = gq; be = bq; }
  else if (which == 1) { g = gk; be = bk; }
  else                 { g = gv; be = bv; }
  float y = (val - mean) * rsqrtf(var + 1e-3f) * g[c] + be[c];

  size_t bh = (size_t)b * HEADS + h;
  if (which == 0) {
    int qrow = (t == 0) ? 0 : 16 + (t - 1);
    Qpad[(bh * QROWS + qrow) * HD + c] = f2bf(y);
    qres[(bh * LPOOL + t) * HD + c] = y;
  } else if (which == 1) {
    int krow = (t == 0) ? 0 : 32 + (t - 1);
    Kpad[(bh * KROWS + krow) * HD + c] = f2bf(y);
  } else {
    int krow = (t == 0) ? 0 : 32 + (t - 1);
    Vt[(bh * HD + c) * KROWS + krow] = f2bf(y);
  }
}

// ---------------------------------------------------------------------------
// rel-pos precompute: rel_h[bh][q][ky] = q_vec . rel_pos_h[qy-ky+31],
//                     rel_w[bh][q][kx] = q_vec . rel_pos_w[qx-kx+31]
// ---------------------------------------------------------------------------
__global__ __launch_bounds__(64)
void relpos_kernel(const float* __restrict__ qres,
                   const float* __restrict__ rel_pos_h, const float* __restrict__ rel_pos_w,
                   float* __restrict__ rel_h, float* __restrict__ rel_w) {
  int blk = blockIdx.x;
  int q = blk % 1024; int bh = blk / 1024;
  int qy = q >> 5, qx = q & 31;
  const float* qv = qres + ((size_t)bh * LPOOL + 1 + q) * HD;
  int tloc = threadIdx.x & 31;
  bool isw = threadIdx.x >= 32;
  int d = isw ? (qx - tloc + 31) : (qy - tloc + 31);
  const float* tab = (isw ? rel_pos_w : rel_pos_h) + (size_t)d * HD;
  float s = 0.f;
#pragma unroll 8
  for (int c = 0; c < HD; ++c) s += qv[c] * tab[c];
  size_t o = ((size_t)bh * 1024 + q) * 32 + tloc;
  if (isw) rel_w[o] = s; else rel_h[o] = s;
}

// ---------------------------------------------------------------------------
// Flash attention: one wave per (bh, 16-query tile). 33 key tiles of 32.
// ---------------------------------------------------------------------------
__global__ __launch_bounds__(32)
void flash_kernel(const u16* __restrict__ Qpad, const u16* __restrict__ Kpad,
                  const u16* __restrict__ Vt, const float* __restrict__ qres,
                  const float* __restrict__ rel_h, const float* __restrict__ rel_w,
                  u16* __restrict__ OutA) {
  int lane = threadIdx.x;
  int qt = blockIdx.x;            // 0..64
  int bh = blockIdx.y;            // 0..47
  const u16* Qb = Qpad + (size_t)bh * QROWS * HD;
  const u16* Kb = Kpad + (size_t)bh * KROWS * HD;
  const u16* Vb = Vt   + (size_t)bh * HD * KROWS;
  const float* rh = rel_h + (size_t)bh * 1024 * 32;
  const float* rw = rel_w + (size_t)bh * 1024 * 32;

  const u16* qrow = Qb + (size_t)(qt * 16 + (lane & 15)) * HD;
  v16bf qa0 = load_fragA(qrow, 0, lane);
  v16bf qa1 = load_fragA(qrow, 32, lane);

  int qidx[8];
  float relw0[8], relw1[8];
#pragma unroll
  for (int r = 0; r < 8; ++r) {
    int m = r + ((lane < 16) ? 0 : 8);
    qidx[r] = (qt - 1) * 16 + m;
    if (qt > 0) {
      relw0[r] = rw[(size_t)qidx[r] * 32 + (lane & 15)];
      relw1[r] = rw[(size_t)qidx[r] * 32 + 16 + (lane & 15)];
    } else {
      relw0[r] = 0.f; relw1[r] = 0.f;
    }
  }

  float mrow[8], lrow[8];
#pragma unroll
  for (int r = 0; r < 8; ++r) { mrow[r] = -1e30f; lrow[r] = 0.f; }
  v8f O[4] = {v8f{}, v8f{}, v8f{}, v8f{}};

  __shared__ __align__(16) u16 Pl[16 * 32];
  const float scale = 0.125f;   // HD^-0.5

  for (int kt = 0; kt < 33; ++kt) {
    // ---- S = Q @ K^T tile (16 x 32) ----
    v8f S0 = v8f{}, S1 = v8f{};
    const u16* k0row = Kb + (size_t)(kt * 32 + (lane & 15)) * HD;
    const u16* k1row = Kb + (size_t)(kt * 32 + 16 + (lane & 15)) * HD;
    v16bf b00 = load_fragB(k0row, 0, lane);
    v16bf b01 = load_fragB(k0row, 32, lane);
    v16bf b10 = load_fragB(k1row, 0, lane);
    v16bf b11 = load_fragB(k1row, 32, lane);
    S0 = __builtin_amdgcn_wmma_f32_16x16x32_bf16(false, qa0, false, b00, (short)0, S0, false, false);
    S0 = __builtin_amdgcn_wmma_f32_16x16x32_bf16(false, qa1, false, b01, (short)0, S0, false, false);
    S1 = __builtin_amdgcn_wmma_f32_16x16x32_bf16(false, qa0, false, b10, (short)0, S1, false, false);
    S1 = __builtin_amdgcn_wmma_f32_16x16x32_bf16(false, qa1, false, b11, (short)0, S1, false, false);

    float relhv[8];
    if (kt > 0 && qt > 0) {
#pragma unroll
      for (int r = 0; r < 8; ++r) relhv[r] = rh[(size_t)qidx[r] * 32 + (kt - 1)];
    } else {
#pragma unroll
      for (int r = 0; r < 8; ++r) relhv[r] = 0.f;
    }

    float p0[8], p1[8];
#pragma unroll
    for (int r = 0; r < 8; ++r) {
      float v0 = S0[r] * scale;
      float v1 = S1[r] * scale;
      if (kt == 0) {
        // only column 0 (cls key) is valid; cls column carries no rel bias
        if ((lane & 15) != 0) v0 = -1e30f;
        v1 = -1e30f;
      } else if (qt > 0) {
        v0 += relhv[r] + relw0[r];
        v1 += relhv[r] + relw1[r];
      }
      // online softmax (row spread across a 16-lane group)
      float tm = fmaxf(v0, v1);
      tm = fmaxf(tm, __shfl_xor(tm, 1, 16));
      tm = fmaxf(tm, __shfl_xor(tm, 2, 16));
      tm = fmaxf(tm, __shfl_xor(tm, 4, 16));
      tm = fmaxf(tm, __shfl_xor(tm, 8, 16));
      float mn = fmaxf(mrow[r], tm);
      float corr = __expf(mrow[r] - mn);
      mrow[r] = mn;
      float e0 = __expf(v0 - mn);
      float e1 = __expf(v1 - mn);
      float rsum = e0 + e1;
      rsum += __shfl_xor(rsum, 1, 16);
      rsum += __shfl_xor(rsum, 2, 16);
      rsum += __shfl_xor(rsum, 4, 16);
      rsum += __shfl_xor(rsum, 8, 16);
      lrow[r] = lrow[r] * corr + rsum;
#pragma unroll
      for (int cg = 0; cg < 4; ++cg) O[cg][r] *= corr;
      p0[r] = e0; p1[r] = e1;
    }

    // ---- repack P (C layout) -> A fragment via LDS ----
#pragma unroll
    for (int r = 0; r < 8; ++r) {
      int m = r + ((lane < 16) ? 0 : 8);
      Pl[m * 32 + (lane & 15)] = f2bf(p0[r]);
      Pl[m * 32 + 16 + (lane & 15)] = f2bf(p1[r]);
    }
    __syncthreads();
    v16bf pa;
    {
      int m = lane & 15;
      int of = (lane < 16) ? 0 : 8;
      FragU f;
      f.u[0] = *reinterpret_cast<const u32x4*>(&Pl[m * 32 + of]);
      f.u[1] = *reinterpret_cast<const u32x4*>(&Pl[m * 32 + of + 16]);
      pa = f.v;
    }
    __syncthreads();

    // ---- O += P @ V tile ----
#pragma unroll
    for (int cg = 0; cg < 4; ++cg) {
      const u16* vrow = Vb + (size_t)(cg * 16 + (lane & 15)) * KROWS;
      v16bf bvf = load_fragB(vrow, kt * 32, lane);
      O[cg] = __builtin_amdgcn_wmma_f32_16x16x32_bf16(false, pa, false, bvf, (short)0, O[cg], false, false);
    }
  }

  // ---- epilogue: out = O / l + q_residual, store bf16 row for proj GEMM ----
  int bb = bh / HEADS, hh = bh - bb * HEADS;
#pragma unroll
  for (int cg = 0; cg < 4; ++cg) {
#pragma unroll
    for (int r = 0; r < 8; ++r) {
      int m = r + ((lane < 16) ? 0 : 8);
      int tok;
      if (qt == 0) { if (m != 0) continue; tok = 0; }
      else tok = 1 + (qt - 1) * 16 + m;
      int col = cg * 16 + (lane & 15);
      float o = O[cg][r] / lrow[r];
      float qv = qres[((size_t)bh * LPOOL + tok) * HD + col];
      OutA[((size_t)(bb * LPOOL + tok)) * DIM + hh * HD + col] = f2bf(o + qv);
    }
  }
}

// ---------------------------------------------------------------------------
// Output projection GEMM: [MPROJ_PAD x 768] @ Bt [768 x 768] + bias -> fp32
// ---------------------------------------------------------------------------
__global__ __launch_bounds__(32)
void gemm_proj_kernel(const u16* __restrict__ A, const u16* __restrict__ Bt,
                      const float* __restrict__ bias, float* __restrict__ out, int Mvalid) {
  int lane = threadIdx.x;
  int m0 = blockIdx.x * 16;
  int n0 = blockIdx.y * 64;
  v8f acc[4] = {v8f{}, v8f{}, v8f{}, v8f{}};
  const u16* arow = A + (size_t)(m0 + (lane & 15)) * DIM;
  for (int k = 0; k < DIM; k += 32) {
    __builtin_prefetch((const void*)(arow + k + 128), 0, 1);
    v16bf a = load_fragA(arow, k, lane);
#pragma unroll
    for (int j = 0; j < 4; ++j) {
      const u16* brow = Bt + (size_t)(n0 + j * 16 + (lane & 15)) * DIM;
      v16bf b = load_fragB(brow, k, lane);
      acc[j] = __builtin_amdgcn_wmma_f32_16x16x32_bf16(false, a, false, b, (short)0, acc[j], false, false);
    }
  }
#pragma unroll
  for (int j = 0; j < 4; ++j) {
#pragma unroll
    for (int r = 0; r < 8; ++r) {
      int row = m0 + r + ((lane < 16) ? 0 : 8);
      if (row >= Mvalid) continue;
      int col = n0 + j * 16 + (lane & 15);
      out[(size_t)row * DIM + col] = acc[j][r] + bias[col];
    }
  }
}

// ---------------------------------------------------------------------------
// Host launcher
// ---------------------------------------------------------------------------
extern "C" void kernel_launch(void* const* d_in, const int* in_sizes, int n_in,
                              void* d_out, int out_size, void* d_ws, size_t ws_size,
                              hipStream_t stream) {
  (void)in_sizes; (void)n_in; (void)out_size; (void)ws_size;
  const float* x      = (const float*)d_in[0];
  const float* w_qkv  = (const float*)d_in[1];
  const float* dw_q   = (const float*)d_in[2];
  const float* dw_k   = (const float*)d_in[3];
  const float* dw_v   = (const float*)d_in[4];
  const float* gq     = (const float*)d_in[5];
  const float* bq     = (const float*)d_in[6];
  const float* gk     = (const float*)d_in[7];
  const float* bk     = (const float*)d_in[8];
  const float* gv     = (const float*)d_in[9];
  const float* bv     = (const float*)d_in[10];
  const float* rph    = (const float*)d_in[11];
  const float* rpw    = (const float*)d_in[12];
  const float* w_proj = (const float*)d_in[13];
  const float* b_proj = (const float*)d_in[14];
  float* out = (float*)d_out;

  char* ws = (char*)d_ws;
  size_t off = 0;
  auto take = [&](size_t bytes) -> size_t {
    size_t o = off; off += (bytes + 255) & ~(size_t)255; return o;
  };
  // zero-initialized region (padded bf16 tiles)
  size_t o_xb    = take((size_t)MQKV_PAD * DIM * 2);
  size_t o_qpad  = take((size_t)48 * QROWS * HD * 2);
  size_t o_kpad  = take((size_t)48 * KROWS * HD * 2);
  size_t o_vt    = take((size_t)48 * HD * KROWS * 2);
  size_t o_outA  = take((size_t)MPROJ_PAD * DIM * 2);
  size_t zero_end = off;
  // rest
  size_t o_wqkvT = take((size_t)2304 * 768 * 2);
  size_t o_wpT   = take((size_t)768 * 768 * 2);
  size_t o_qkv   = take((size_t)3 * 48 * NTOK * HD * 2);
  size_t o_qres  = take((size_t)48 * LPOOL * HD * 4);
  size_t o_relh  = take((size_t)48 * 1024 * 32 * 4);
  size_t o_relw  = take((size_t)48 * 1024 * 32 * 4);

  u16*   xb    = (u16*)(ws + o_xb);
  u16*   Qpad  = (u16*)(ws + o_qpad);
  u16*   Kpad  = (u16*)(ws + o_kpad);
  u16*   Vt    = (u16*)(ws + o_vt);
  u16*   OutA  = (u16*)(ws + o_outA);
  u16*   wqkvT = (u16*)(ws + o_wqkvT);
  u16*   wpT   = (u16*)(ws + o_wpT);
  u16*   qkvb  = (u16*)(ws + o_qkv);
  float* qres  = (float*)(ws + o_qres);
  float* relh  = (float*)(ws + o_relh);
  float* relw  = (float*)(ws + o_relw);

  hipMemsetAsync(ws, 0, zero_end, stream);

  { long long n = (long long)MQKV * DIM;
    f32_to_bf16_kernel<<<(unsigned)((n + 255) / 256), 256, 0, stream>>>(x, xb, n); }
  { long long n = 2304LL * 768;
    transpose_bf16_kernel<<<(unsigned)((n + 255) / 256), 256, 0, stream>>>(w_qkv, wqkvT, 768, 2304); }
  { long long n = 768LL * 768;
    transpose_bf16_kernel<<<(unsigned)((n + 255) / 256), 256, 0, stream>>>(w_proj, wpT, 768, 768); }

  gemm_qkv_kernel<<<dim3(MQKV_PAD / 16, 2304 / 64), 32, 0, stream>>>(xb, wqkvT, qkvb, MQKV);

  pool_ln_kernel<<<3 * BATCH * HEADS * LPOOL, 64, 0, stream>>>(
      qkvb, dw_q, dw_k, dw_v, gq, bq, gk, bk, gv, bv, Qpad, qres, Kpad, Vt);

  relpos_kernel<<<48 * 1024, 64, 0, stream>>>(qres, rph, rpw, relh, relw);

  flash_kernel<<<dim3(65, 48), 32, 0, stream>>>(Qpad, Kpad, Vt, qres, relh, relw, OutA);

  gemm_proj_kernel<<<dim3(MPROJ_PAD / 16, 768 / 64), 32, 0, stream>>>(OutA, wpT, b_proj, out, MPROJ);
}
